// Discriminator_90546500534752
// MI455X (gfx1250) — compile-verified, matrix operand order checked by reading, not run on hardware
//
#include <hip/hip_runtime.h>
#include <hip/hip_bf16.h>
#include <cstdint>

typedef __attribute__((ext_vector_type(16))) _Float16 v16h;
typedef __attribute__((ext_vector_type(8)))  _Float16 v8h;
typedef __attribute__((ext_vector_type(8)))  float    v8f;

// ---------------------------------------------------------------------------
// ws layout (folded weights), computed by prep_kernel each launch:
//   _Float16 KsT[128*32]  transposed: KsT[f][k]; rows of K: k 0..9 = (W1@W2)@Ws_top,
//                         k 16..25 = Ws_bot, rest 0
//   _Float16 KtT[128*32]
//   float    vs1[128], vs2[128], bsv[128], vt1[128], vt2[128], btv[128]
//   float    wH[144], cHead
// ---------------------------------------------------------------------------

__global__ __launch_bounds__(256) void prep_kernel(
    const float* __restrict__ W1, const float* __restrict__ b1,
    const float* __restrict__ W2, const float* __restrict__ b2,
    const float* __restrict__ Ws, const float* __restrict__ bs,
    const float* __restrict__ Wt, const float* __restrict__ bt,
    const float* __restrict__ Wl1, const float* __restrict__ bl1,
    const float* __restrict__ Wl2, const float* __restrict__ bl2,
    const float* __restrict__ Wo, const float* __restrict__ bo,
    void* __restrict__ ws)
{
    __shared__ float W12[10][64];   // W1 @ W2
    __shared__ float b12s[64];      // b1 @ W2
    __shared__ float t1s[128];      // Wl2 @ Wo

    _Float16* KsT = (_Float16*)ws;
    _Float16* KtT = KsT + 128*32;
    float* fb   = (float*)(KtT + 128*32);
    float* vs1  = fb;       float* vs2 = fb+128; float* bsv = fb+256;
    float* vt1  = fb+384;   float* vt2 = fb+512; float* btv = fb+640;
    float* wH   = fb+768;   float* cHead = fb+912;

    const int t = threadIdx.x;

    for (int idx = t; idx < 640; idx += 256) {
        int d = idx >> 6, g = idx & 63;
        float acc = 0.f;
        for (int k = 0; k < 128; ++k) acc += W1[d*128+k]*W2[k*64+g];
        W12[d][g] = acc;
    }
    for (int g = t; g < 64; g += 256) {
        float acc = 0.f;
        for (int k = 0; k < 128; ++k) acc += b1[k]*W2[k*64+g];
        b12s[g] = acc;
    }
    for (int k = t; k < 128; k += 256) {
        float acc = 0.f;
        for (int j = 0; j < 64; ++j) acc += Wl2[k*64+j]*Wo[j];
        t1s[k] = acc;
    }
    __syncthreads();

    // KsT / KtT : transposed [f][k], K padded to 32 for the f16 WMMA (K=32)
    for (int idx = t; idx < 128*32; idx += 256) {
        int f = idx >> 5, k = idx & 31;
        float vS = 0.f, vT = 0.f;
        if (k < 10) {
            for (int g = 0; g < 64; ++g) {
                vS += W12[k][g]*Ws[g*128+f];
                vT += W12[k][g]*Wt[g*128+f];
            }
        } else if (k >= 16 && k < 26) {
            vS = Ws[(64 + k - 16)*128 + f];
            vT = Wt[(64 + k - 16)*128 + f];
        }
        KsT[idx] = (_Float16)vS;
        KtT[idx] = (_Float16)vT;
    }
    for (int f = t; f < 128; f += 256) {
        float a2 = 0.f, a1 = 0.f, c2 = 0.f, c1 = 0.f;
        for (int g = 0; g < 64; ++g) {
            a2 += b12s[g]*Ws[g*128+f];  a1 += b2[g]*Ws[g*128+f];
            c2 += b12s[g]*Wt[g*128+f];  c1 += b2[g]*Wt[g*128+f];
        }
        vs2[f] = a2; vs1[f] = a1; bsv[f] = bs[f];
        vt2[f] = c2; vt1[f] = c1; btv[f] = bt[f];
    }
    for (int z = t; z < 144; z += 256) {
        float acc = 0.f;
        for (int k = 0; k < 128; ++k) acc += Wl1[z*128+k]*t1s[k];
        wH[z] = acc;
    }
    if (t == 0) {
        float acc = bo[0];
        for (int k = 0; k < 128; ++k) acc += bl1[k]*t1s[k];
        for (int j = 0; j < 64;  ++j) acc += bl2[j]*Wo[j];
        *cHead = acc;
    }
}

// --- wave32 WMMA f16 16x16x32 fragment loader (ISA 7.12.2) -----------------
// Per-lane data is two contiguous 8-half runs in the K dimension:
//   elems 0..7  <-> K = 8*half .. 8*half+7
//   elems 8..15 <-> K = 16+8*half .. 16+8*half+7
// With a K-contiguous row-major tile (ld halves per row, 16B-aligned rows)
// this is exactly two ds_load_b128 per lane. Works for A tiles (row = M) and
// for B tiles stored transposed (row = N).
__device__ inline v16h frag_ld(const _Float16* __restrict__ src, int ld, int rbase, int lane) {
    const int half = (lane >> 4) & 1;
    const _Float16* row = src + (rbase + (lane & 15)) * ld + 8 * half;
    v8h lo = *(const v8h*)(row);
    v8h hi = *(const v8h*)(row + 16);
    v16h r;
#pragma unroll
    for (int i = 0; i < 8; ++i) { r[i] = lo[i]; r[8+i] = hi[i]; }
    return r;
}

__device__ inline float sigmoidf_(float x) { return 1.0f / (1.0f + __expf(-x)); }

// ---------------------------------------------------------------------------
// main kernel: one workgroup (8 wave32) per graph
// ---------------------------------------------------------------------------
__global__ __launch_bounds__(256) void molgan_disc_kernel(
    const float* __restrict__ adjg, const float* __restrict__ nodesg,
    const float* __restrict__ We,   const float* __restrict__ be,
    const void* __restrict__ ws,    float* __restrict__ out)
{
    __shared__ float ch[4][32][32];            // per-edge-type adjacency
    __shared__ float Asum_f[32][32];
    __shared__ __align__(16) _Float16 Ah[32][32];       // Asum + I (A tile)
    __shared__ float colsum[4][32];
    __shared__ float r1s[32], r2s[32], Ssum[4];
    __shared__ float uu[4][10];
    __shared__ float heL[16];
    __shared__ float nodes_f[32][10];
    __shared__ __align__(16) _Float16 nodes_hT[16][32]; // B tile, transposed [d][j]
    __shared__ __align__(16) _Float16 X1hT[16][32];     // B tile, transposed [d][j]
    __shared__ __align__(16) _Float16 Gh[32][32];       // A tile [node][feat]: [X2|nodes|0]
    __shared__ __align__(16) _Float16 KslT[128][32];    // B tile, transposed [f][k]
    __shared__ __align__(16) _Float16 KtlT[128][32];
    __shared__ float vs1L[128], vs2L[128], bsL[128];
    __shared__ float vt1L[128], vt2L[128], btL[128];
    __shared__ float wHL[144];
    __shared__ float hgsum[128];
    __shared__ float red[144];

    const int t    = threadIdx.x;
    const int b    = blockIdx.x;
    const int lane = t & 31;
    const int w    = t >> 5;

    const _Float16* KsT = (const _Float16*)ws;
    const _Float16* KtT = KsT + 128*32;
    const float*    fb  = (const float*)(KtT + 128*32);

    // ---------------- phase 0: stream adj (coalesced b128), stage weights ---
    {
        const float4* src4 = (const float4*)(adjg + (size_t)b * 5120);
        float4 q[5];
#pragma unroll
        for (int x = 0; x < 5; ++x) q[x] = src4[5*t + x];
        float vals[20];
#pragma unroll
        for (int x = 0; x < 5; ++x) {
            vals[4*x+0] = q[x].x; vals[4*x+1] = q[x].y;
            vals[4*x+2] = q[x].z; vals[4*x+3] = q[x].w;
        }
#pragma unroll
        for (int k = 0; k < 4; ++k) {          // pairs 4t..4t+3, 5 channels each
            int p = 4*t + k, i = p >> 5, j = p & 31;
            float a1 = vals[5*k+1], a2 = vals[5*k+2], a3 = vals[5*k+3], a4 = vals[5*k+4];
            ch[0][i][j] = a1; ch[1][i][j] = a2; ch[2][i][j] = a3; ch[3][i][j] = a4;
            float s = a1 + a2 + a3 + a4;
            Asum_f[i][j] = s;
            Ah[i][j] = (_Float16)(s + ((i == j) ? 1.0f : 0.0f));
        }
        const float* nb = nodesg + (size_t)b * 320;
        for (int idx = t; idx < 320; idx += 256) {
            int j = idx / 10, d = idx - j*10;
            nodes_f[j][d] = nb[idx];
        }
        for (int idx = t; idx < 512; idx += 256) {     // nodes_hT[d][j], zero-padded d>=10
            int d = idx >> 5, j = idx & 31;
            float v = (d < 10) ? nb[j*10 + d] : 0.0f;
            nodes_hT[d][j] = (_Float16)v;
        }
        for (int idx = t; idx < 512; idx += 256) {     // Gh cols 16..31 (nodes | 0)
            int j = idx >> 4, c = 16 + (idx & 15);
            float v = (c < 26) ? nb[j*10 + (c - 16)] : 0.0f;
            Gh[j][c] = (_Float16)v;
        }
        const uint32_t* ksw = (const uint32_t*)KsT;
        const uint32_t* ktw = (const uint32_t*)KtT;
        uint32_t* kslw = (uint32_t*)&KslT[0][0];
        uint32_t* ktlw = (uint32_t*)&KtlT[0][0];
        for (int idx = t; idx < 2048; idx += 256) { kslw[idx] = ksw[idx]; ktlw[idx] = ktw[idx]; }
        for (int f = t; f < 128; f += 256) {
            vs1L[f] = fb[f];       vs2L[f] = fb[128+f]; bsL[f] = fb[256+f];
            vt1L[f] = fb[384+f];   vt2L[f] = fb[512+f]; btL[f] = fb[640+f];
        }
        for (int z = t; z < 144; z += 256) wHL[z] = fb[768+z];
    }
    __syncthreads();   // S0

    // ---------------- phase 1: colsum, r1, X1 = (Asum+I)@nodes (WMMA) ------
    if (t < 128) {                       // colsum[e][j]
        int e = t >> 5, j = t & 31;
        float acc = 0.f;
        for (int i = 0; i < 32; ++i) acc += ch[e][i][j];
        colsum[e][j] = acc;
    } else if (t < 160) {                // r1[i] = 1 + rowsum(Asum)
        int i = t - 128;
        float acc = 1.0f;
        for (int j = 0; j < 32; ++j) acc += Asum_f[i][j];
        r1s[i] = acc;
    } else if (w >= 6) {                 // waves 6,7: X1 tiles (mt = w-6)
        int mt = w - 6;
        v16h a  = frag_ld(&Ah[0][0], 32, mt*16, lane);
        v16h bf = frag_ld(&nodes_hT[0][0], 32, 0, lane);
        v8f  c  = {};
        c = __builtin_amdgcn_wmma_f32_16x16x32_f16(false, a, false, bf, (short)0, c, false, false);
        const int n = lane & 15, half = (lane >> 4) & 1;
        v8h ph;
#pragma unroll
        for (int r = 0; r < 8; ++r) ph[r] = (_Float16)c[r];
        // lane's 8 results are consecutive m in row n -> one b128 store
        *(v8h*)(&X1hT[n][mt*16 + 8*half]) = ph;
    }
    __syncthreads();   // S1

    // ---------------- phase 2: r2, S_e, u_e, X2 = (Asum+I)@X1 (WMMA) -------
    if (t < 32) {
        float acc = r1s[t];
        for (int j = 0; j < 32; ++j) acc += Asum_f[t][j] * r1s[j];
        r2s[t] = acc;
    } else if (t < 36) {
        int e = t - 32;
        float acc = 0.f;
        for (int j = 0; j < 32; ++j) acc += colsum[e][j];
        Ssum[e] = acc;
    } else if (t >= 64 && t < 104) {
        int q = t - 64, e = q / 10, d = q - e*10;
        float acc = 0.f;
        for (int j = 0; j < 32; ++j) acc += colsum[e][j] * nodes_f[j][d];
        uu[e][d] = acc;
    } else if (w >= 6) {                 // waves 6,7: X2 tiles -> Gh cols 0..15
        int mt = w - 6;
        v16h a  = frag_ld(&Ah[0][0], 32, mt*16, lane);
        v16h bf = frag_ld(&X1hT[0][0], 32, 0, lane);
        v8f  c  = {};
        c = __builtin_amdgcn_wmma_f32_16x16x32_f16(false, a, false, bf, (short)0, c, false, false);
        const int n = lane & 15, half = (lane >> 4) & 1;
#pragma unroll
        for (int r = 0; r < 8; ++r)
            Gh[mt*16 + r + 8*half][n] = (_Float16)c[r];
    }
    __syncthreads();   // S2

    // ---------------- phase 3: gates (4 WMMAs per wave), column reduce -----
    {
        // wave w owns output columns f in [16w, 16w+16)
        v16h a0  = frag_ld(&Gh[0][0], 32, 0,  lane);
        v16h a1  = frag_ld(&Gh[0][0], 32, 16, lane);
        v16h bfs = frag_ld(&KslT[0][0], 32, w*16, lane);
        v16h bft = frag_ld(&KtlT[0][0], 32, w*16, lane);
        v8f z = {};
        v8f s0 = __builtin_amdgcn_wmma_f32_16x16x32_f16(false, a0, false, bfs, (short)0, z, false, false);
        v8f s1 = __builtin_amdgcn_wmma_f32_16x16x32_f16(false, a1, false, bfs, (short)0, z, false, false);
        v8f t0 = __builtin_amdgcn_wmma_f32_16x16x32_f16(false, a0, false, bft, (short)0, z, false, false);
        v8f t1 = __builtin_amdgcn_wmma_f32_16x16x32_f16(false, a1, false, bft, (short)0, z, false, false);

        const int n = lane & 15, half = (lane >> 4) & 1;
        const int f = w*16 + n;
        const float cs2 = vs2L[f], cs1 = vs1L[f], cbs = bsL[f];
        const float ct2 = vt2L[f], ct1 = vt1L[f], cbt = btL[f];
        float colacc = 0.f;
#pragma unroll
        for (int r = 0; r < 8; ++r) {
            int m0 = r + 8*half;
            int m1 = 16 + m0;
            float sv0 = s0[r] + r2s[m0]*cs2 + r1s[m0]*cs1 + cbs;
            float tv0 = t0[r] + r2s[m0]*ct2 + r1s[m0]*ct1 + cbt;
            float sv1 = s1[r] + r2s[m1]*cs2 + r1s[m1]*cs1 + cbs;
            float tv1 = t1[r] + r2s[m1]*ct2 + r1s[m1]*ct1 + cbt;
            colacc += sigmoidf_(sv0)*tanhf(tv0) + sigmoidf_(sv1)*tanhf(tv1);
        }
        colacc += __shfl_xor(colacc, 16, 32);
        if (lane < 16) hgsum[f] = colacc;

        // edge-conv output (wave 0 tail): he = tanh(sum_e u_e@We_e + S_e*be_e)
        if (w == 0 && lane < 16) {
            int fe = lane;
            float acc = 0.f;
            for (int e = 0; e < 4; ++e) {
                float dot = 0.f;
                for (int d = 0; d < 10; ++d) dot += uu[e][d] * We[(e*10 + d)*16 + fe];
                acc += dot + Ssum[e] * be[e*16 + fe];
            }
            heL[fe] = tanhf(acc);
        }
    }
    __syncthreads();   // S3

    // ---------------- phase 4: folded head ---------------------------------
    if (t < 144) {
        float v = (t < 128) ? tanhf(hgsum[t]) * wHL[t] : heL[t - 128] * wHL[t];
        red[t] = v;
    }
    __syncthreads();
    if (t == 0) {
        float acc = fb[912];             // cHead
        for (int z2 = 0; z2 < 144; ++z2) acc += red[z2];
        out[b] = acc;
    }
}

extern "C" void kernel_launch(void* const* d_in, const int* in_sizes, int n_in,
                              void* d_out, int out_size, void* d_ws, size_t ws_size,
                              hipStream_t stream)
{
    const float* adj  = (const float*)d_in[0];
    const float* nod  = (const float*)d_in[1];
    const float* W1   = (const float*)d_in[2];
    const float* b1   = (const float*)d_in[3];
    const float* W2   = (const float*)d_in[4];
    const float* b2   = (const float*)d_in[5];
    const float* We   = (const float*)d_in[6];
    const float* be   = (const float*)d_in[7];
    const float* Ws   = (const float*)d_in[8];
    const float* bs   = (const float*)d_in[9];
    const float* Wt   = (const float*)d_in[10];
    const float* bt   = (const float*)d_in[11];
    const float* Wl1  = (const float*)d_in[12];
    const float* bl1  = (const float*)d_in[13];
    const float* Wl2  = (const float*)d_in[14];
    const float* bl2  = (const float*)d_in[15];
    const float* Wo   = (const float*)d_in[16];
    const float* bo   = (const float*)d_in[17];

    prep_kernel<<<1, 256, 0, stream>>>(W1, b1, W2, b2, Ws, bs, Wt, bt,
                                       Wl1, bl1, Wl2, bl2, Wo, bo, d_ws);
    molgan_disc_kernel<<<4096, 256, 0, stream>>>(adj, nod, We, be, d_ws, (float*)d_out);
}
